// Seq2Seq_80599356276770
// MI455X (gfx1250) — matrix operand order repaired and checked by reading, not compiled, not run on hardware
//
#include <hip/hip_runtime.h>
#include <hip/hip_bf16.h>

// ---------------------------------------------------------------------------
// Seq2Seq 3-layer LSTM encoder/decoder for MI455X (gfx1250, wave32, WMMA).
//
// Batch-parallel persistent workgroups: each WG owns BM=32 batch rows and
// runs all T=256 encoder + T=256 decoder steps with h/c/y state, ALL layer
// weights (f16) and the x input slice resident in the 320KB WGP LDS.
// Per step+layer: [32x64] @ [64x256] GEMM via v_wmma_f32_16x16x32_f16,
// 8 waves tiled 2(M) x 4(N), f32 accumulation. Gate nonlinearities use the
// CDNA5 native TANH / EXP / RCP transcendentals (branch-free).
// ---------------------------------------------------------------------------

typedef __attribute__((ext_vector_type(16))) _Float16 v16h;
typedef __attribute__((ext_vector_type(8)))  _Float16 v8h;
typedef __attribute__((ext_vector_type(8)))  float    v8f;

#define BATCH    1024
#define SEQ_T    256
#define HID      64
#define NGATE    256          // 4*H
#define BM       32           // batch rows per workgroup
#define NTHREADS 256          // 8 waves (wave32)
#define WSZ      (NGATE*HID)  // one weight matrix, elements

static_assert((BM * HID) % NTHREADS == 0, "elementwise tiling");

// --- branch-free transcendentals on the gfx1250 TRANS pipe ------------------
#if __has_builtin(__builtin_amdgcn_exp2f)
#define EXP2F(x) __builtin_amdgcn_exp2f(x)
#else
#define EXP2F(x) exp2f(x)
#endif
#if __has_builtin(__builtin_amdgcn_rcpf)
#define RCPF(x) __builtin_amdgcn_rcpf(x)
#else
#define RCPF(x) (1.0f / (x))
#endif

__device__ __forceinline__ float sigm(float x) {
  // 1 / (1 + 2^(-x*log2e)) : v_exp_f32 + v_rcp_f32, no branches
  return RCPF(1.0f + EXP2F(x * -1.4426950408889634f));
}
__device__ __forceinline__ float fast_tanh(float x) {
#if __has_builtin(__builtin_amdgcn_tanhf)
  return __builtin_amdgcn_tanhf(x);      // native v_tanh_f32 (gfx1250 TRANS op)
#else
  // tanh(x) = 1 - 2/(2^(2x*log2e) + 1), branch-free
  return 1.0f - 2.0f * RCPF(EXP2F(x * 2.8853900817779268f) + 1.0f);
#endif
}

// --- WMMA fragment loads from LDS, per CDNA5 ISA wave32 layouts -------------
// A: 16x32 f16. lane L -> row M=L&15; halves 0-7 = A[M][kb..kb+7],
// halves 8-15 = A[M][kb+16..kb+23], kb = (L>>4)*8 (+32 per k-chunk).
__device__ __forceinline__ v16h ldA(const _Float16* __restrict__ h, int m, int kc, int lane) {
  const int kh = kc * 32 + ((lane >> 4) << 3);
  v8h lo = *(const v8h*)(h + m * HID + kh);
  v8h hi = *(const v8h*)(h + m * HID + kh + 16);
  return __builtin_shufflevector(lo, hi, 0,1,2,3,4,5,6,7,8,9,10,11,12,13,14,15);
}
// B: 32x16 f16 (K x N). lane L -> col N=L&15; halves 0-7 = B[kb..kb+7][N],
// halves 8-15 = B[kb+16..kb+23][N], kb = (L>>4)*8 (+32 per k-chunk).
// Weights stored LDS[n][k] row-major, so each 8-half group is 16 contiguous bytes.
__device__ __forceinline__ v16h ldB(const _Float16* __restrict__ W, int n, int kc, int lane) {
  const int kh = kc * 32 + ((lane >> 4) << 3);
  v8h lo = *(const v8h*)(W + n * HID + kh);
  v8h hi = *(const v8h*)(W + n * HID + kh + 16);
  return __builtin_shufflevector(lo, hi, 0,1,2,3,4,5,6,7,8,9,10,11,12,13,14,15);
}

// Accumulate [32x64] @ [64x256] -> this wave's 16x64 C stripe (4 tiles).
__device__ __forceinline__ void gemm_acc(const _Float16* __restrict__ A,
                                         const _Float16* __restrict__ W,
                                         int r0, int n0, int lane, v8f acc[4]) {
#pragma unroll
  for (int kc = 0; kc < 2; ++kc) {          // K = 64 in two K=32 chunks
    v16h a = ldA(A, r0 + (lane & 15), kc, lane);
#pragma unroll
    for (int t = 0; t < 4; ++t) {
      v16h b = ldB(W, n0 + 16 * t + (lane & 15), kc, lane);
      acc[t] = __builtin_amdgcn_wmma_f32_16x16x32_f16(
          /*neg_a=*/false, a, /*neg_b=*/false, b,
          /*c_mod=*/(short)0, acc[t], /*reuse_a=*/false, /*reuse_b=*/false);
    }
  }
}

// C/D layout: lane L -> col N=L&15; VGPR r -> row r (+8 for lanes 16-31).
__device__ __forceinline__ void store_z(float* __restrict__ z, int r0, int n0,
                                        int lane, const v8f acc[4]) {
  const int rb = r0 + ((lane >> 4) << 3);
  const int c0 = lane & 15;
#pragma unroll
  for (int t = 0; t < 4; ++t)
#pragma unroll
    for (int r = 0; r < 8; ++r)
      z[(rb + r) * NGATE + n0 + 16 * t + c0] = acc[t][r];
}

// Gate nonlinearities + state update. Adds bias and the rank-1 input term
// (x[b] * Wih[:,0]) when hasx (encoder/decoder layer 0, since I==1).
__device__ __forceinline__ void lstm_elem(const float* __restrict__ z,
                                          const float* __restrict__ bias,
                                          const float* __restrict__ wx,
                                          const float* __restrict__ xcol,  // xbuf + t
                                          float* __restrict__ c,
                                          _Float16* __restrict__ h,
                                          int tid, bool hasx) {
#pragma unroll
  for (int e = 0; e < (BM * HID) / NTHREADS; ++e) {
    const int idx = tid + e * NTHREADS;
    const int row = idx >> 6, k = idx & 63;
    float zi = z[row * NGATE + k]        + bias[k];
    float zf = z[row * NGATE + 64 + k]   + bias[64 + k];
    float zg = z[row * NGATE + 128 + k]  + bias[128 + k];
    float zo = z[row * NGATE + 192 + k]  + bias[192 + k];
    if (hasx) {
      const float xv = xcol[row * SEQ_T];
      zi += xv * wx[k];       zf += xv * wx[64 + k];
      zg += xv * wx[128 + k]; zo += xv * wx[192 + k];
    }
    const float ig = sigm(zi), fg = sigm(zf), og = sigm(zo);
    const float gg = fast_tanh(zg);
    const float cn = fg * c[idx] + ig * gg;
    c[idx] = cn;
    h[idx] = (_Float16)(og * fast_tanh(cn));
  }
}

// One LSTM layer step: z = A0@W0^T (+ A1@W1^T) -> gates -> update h/c.
__device__ __forceinline__ void lstm_layer(const _Float16* __restrict__ A0,
                                           const _Float16* __restrict__ W0,
                                           const _Float16* __restrict__ A1,
                                           const _Float16* __restrict__ W1,
                                           float* __restrict__ zS,
                                           const float* __restrict__ bias,
                                           const float* __restrict__ wx,
                                           const float* __restrict__ xcol,
                                           float* __restrict__ c,
                                           _Float16* __restrict__ h,
                                           int tid, int lane, int r0, int n0,
                                           bool hasx) {
  v8f zero = {0.f, 0.f, 0.f, 0.f, 0.f, 0.f, 0.f, 0.f};
  v8f acc[4] = {zero, zero, zero, zero};
  gemm_acc(A0, W0, r0, n0, lane, acc);
  if (A1) gemm_acc(A1, W1, r0, n0, lane, acc);
  store_z(zS, r0, n0, lane, acc);
  __syncthreads();
  lstm_elem(zS, bias, wx, xcol, c, h, tid, hasx);
  __syncthreads();
}

__device__ __forceinline__ void load_w(const float* __restrict__ g,
                                       _Float16* __restrict__ d, int tid) {
  for (int i = tid; i < WSZ; i += NTHREADS) d[i] = (_Float16)g[i];
}

__global__ __launch_bounds__(NTHREADS, 1)
void seq2seq_lstm_kernel(const float* __restrict__ enc_x, const float* __restrict__ dec_x,
                         const float* __restrict__ eWih0, const float* __restrict__ eWhh0, const float* __restrict__ eB0,
                         const float* __restrict__ eWih1, const float* __restrict__ eWhh1, const float* __restrict__ eB1,
                         const float* __restrict__ eWih2, const float* __restrict__ eWhh2, const float* __restrict__ eB2,
                         const float* __restrict__ dWih0, const float* __restrict__ dWhh0, const float* __restrict__ dB0,
                         const float* __restrict__ dWih1, const float* __restrict__ dWhh1, const float* __restrict__ dB1,
                         const float* __restrict__ dWih2, const float* __restrict__ dWhh2, const float* __restrict__ dB2,
                         const float* __restrict__ fcW, const float* __restrict__ fcB,
                         float* __restrict__ out) {
  extern __shared__ char smem[];
  _Float16* Wlds = (_Float16*)smem;                                        // 6 * 32KB
  float*    cS   = (float*)(smem + (size_t)6 * WSZ * 2);                   // 3*BM*HID f32
  _Float16* hS   = (_Float16*)((char*)cS + (size_t)3 * BM * HID * 4);      // 3*BM*HID f16
  _Float16* yS   = hS + 3 * BM * HID;                                      // BM*HID f16
  float*    zS   = (float*)((char*)(yS + BM * HID));                       // BM*NGATE f32
  float*    bS   = zS + BM * NGATE;                                        // 6*NGATE f32
  float*    wxE  = bS + 6 * NGATE;                                         // NGATE
  float*    wxD  = wxE + NGATE;                                            // NGATE
  float*    fcWs = wxD + NGATE;                                            // HID
  float*    xbuf = fcWs + HID;                                             // BM*SEQ_T (32KB)

  const int tid  = threadIdx.x;
  const int lane = tid & 31;
  const int wave = tid >> 5;
  const int r0 = (wave & 1) * 16;   // wave's M tile base (0 or 16)
  const int n0 = (wave >> 1) * 64;  // wave's N stripe base (0/64/128/192)
  const int rowBase = blockIdx.x * BM;

  // ---- init state + small constants ----
  for (int i = tid; i < 3 * BM * HID; i += NTHREADS) { cS[i] = 0.f; hS[i] = (_Float16)0.f; }
  for (int i = tid; i < BM * HID; i += NTHREADS) yS[i] = (_Float16)0.f;
  for (int i = tid; i < NGATE; i += NTHREADS) {
    bS[i] = eB0[i]; bS[NGATE + i] = eB1[i]; bS[2 * NGATE + i] = eB2[i];
    bS[3 * NGATE + i] = dB0[i]; bS[4 * NGATE + i] = dB1[i]; bS[5 * NGATE + i] = dB2[i];
    wxE[i] = eWih0[i];          // e_Wih0 is [256][1]
    wxD[i] = dWih0[i * 65];     // column 0 of d_Wih0 [256][65]
  }
  if (tid < HID) fcWs[tid] = fcW[tid];
  const float fcb = fcB[0];

  // ---- stage this WG's encoder x slice (coalesced, once) ----
  for (int i = tid; i < BM * SEQ_T; i += NTHREADS)
    xbuf[i] = enc_x[(size_t)rowBase * SEQ_T + i];

  // ---- encoder weights -> LDS f16 (slots 1..5; slot 0 unused in enc) ----
  load_w(eWhh0, Wlds + 1 * WSZ, tid);
  load_w(eWih1, Wlds + 2 * WSZ, tid);
  load_w(eWhh1, Wlds + 3 * WSZ, tid);
  load_w(eWih2, Wlds + 4 * WSZ, tid);
  load_w(eWhh2, Wlds + 5 * WSZ, tid);
  __syncthreads();

  // ---- encoder: T steps, all operands in LDS ----
  for (int t = 0; t < SEQ_T; ++t) {
    lstm_layer(hS, Wlds + 1 * WSZ, nullptr, nullptr,
               zS, bS, wxE, xbuf + t, cS, hS, tid, lane, r0, n0, true);
    lstm_layer(hS, Wlds + 2 * WSZ, hS + BM * HID, Wlds + 3 * WSZ,
               zS, bS + NGATE, nullptr, nullptr,
               cS + BM * HID, hS + BM * HID, tid, lane, r0, n0, false);
    lstm_layer(hS + BM * HID, Wlds + 4 * WSZ, hS + 2 * BM * HID, Wlds + 5 * WSZ,
               zS, bS + 2 * NGATE, nullptr, nullptr,
               cS + 2 * BM * HID, hS + 2 * BM * HID, tid, lane, r0, n0, false);
  }

  // ---- swap in decoder weights + x slice (enc drained by trailing barrier) ----
  for (int i = tid; i < WSZ; i += NTHREADS) {         // d_Wih0 cols 1..64
    const int n = i >> 6, k = i & 63;
    Wlds[i] = (_Float16)dWih0[n * 65 + 1 + k];
  }
  load_w(dWhh0, Wlds + 1 * WSZ, tid);
  load_w(dWih1, Wlds + 2 * WSZ, tid);
  load_w(dWhh1, Wlds + 3 * WSZ, tid);
  load_w(dWih2, Wlds + 4 * WSZ, tid);
  load_w(dWhh2, Wlds + 5 * WSZ, tid);
  for (int i = tid; i < BM * SEQ_T; i += NTHREADS)
    xbuf[i] = dec_x[(size_t)rowBase * SEQ_T + i];
  __syncthreads();

  // ---- decoder: input = concat(x_t, y_{t-1}); y_0 = 0, h/c carry over ----
  for (int t = 0; t < SEQ_T; ++t) {
    lstm_layer(yS, Wlds, hS, Wlds + 1 * WSZ,
               zS, bS + 3 * NGATE, wxD, xbuf + t, cS, hS, tid, lane, r0, n0, true);
    lstm_layer(hS, Wlds + 2 * WSZ, hS + BM * HID, Wlds + 3 * WSZ,
               zS, bS + 4 * NGATE, nullptr, nullptr,
               cS + BM * HID, hS + BM * HID, tid, lane, r0, n0, false);
    lstm_layer(hS + BM * HID, Wlds + 4 * WSZ, hS + 2 * BM * HID, Wlds + 5 * WSZ,
               zS, bS + 5 * NGATE, nullptr, nullptr,
               cS + 2 * BM * HID, hS + 2 * BM * HID, tid, lane, r0, n0, false);

    // y <- new top-layer h
    for (int i = tid; i < BM * HID; i += NTHREADS) yS[i] = hS[2 * BM * HID + i];

    // FC head: pred[b] = h2[b,:] . fcW + fcb   (8 lanes per row, shfl reduce)
    {
      const int row = tid >> 3, j = tid & 7;
      float s = 0.f;
#pragma unroll
      for (int k = 0; k < 8; ++k)
        s += (float)hS[2 * BM * HID + row * HID + j * 8 + k] * fcWs[j * 8 + k];
      s += __shfl_xor(s, 1, 32);
      s += __shfl_xor(s, 2, 32);
      s += __shfl_xor(s, 4, 32);
      if (j == 0) out[(size_t)(rowBase + row) * SEQ_T + t] = s + fcb;
    }
    __syncthreads();  // yS ready for next step's layer-0 GEMM
  }
}

// ---------------------------------------------------------------------------
extern "C" void kernel_launch(void* const* d_in, const int* in_sizes, int n_in,
                              void* d_out, int out_size, void* d_ws, size_t ws_size,
                              hipStream_t stream) {
  (void)in_sizes; (void)n_in; (void)out_size; (void)d_ws; (void)ws_size;
  const float* enc_x = (const float*)d_in[0];
  const float* dec_x = (const float*)d_in[1];
  const float* eWih0 = (const float*)d_in[2];
  const float* eWhh0 = (const float*)d_in[3];
  const float* eB0   = (const float*)d_in[4];
  const float* eWih1 = (const float*)d_in[5];
  const float* eWhh1 = (const float*)d_in[6];
  const float* eB1   = (const float*)d_in[7];
  const float* eWih2 = (const float*)d_in[8];
  const float* eWhh2 = (const float*)d_in[9];
  const float* eB2   = (const float*)d_in[10];
  const float* dWih0 = (const float*)d_in[11];
  const float* dWhh0 = (const float*)d_in[12];
  const float* dB0   = (const float*)d_in[13];
  const float* dWih1 = (const float*)d_in[14];
  const float* dWhh1 = (const float*)d_in[15];
  const float* dB1   = (const float*)d_in[16];
  const float* dWih2 = (const float*)d_in[17];
  const float* dWhh2 = (const float*)d_in[18];
  const float* dB2   = (const float*)d_in[19];
  const float* fcW   = (const float*)d_in[20];
  const float* fcB   = (const float*)d_in[21];
  float* out = (float*)d_out;

  constexpr size_t SMEM_BYTES =
      (size_t)6 * WSZ * 2            // f16 weight slots (192 KB)
    + (size_t)3 * BM * HID * 4       // c state (f32)
    + (size_t)3 * BM * HID * 2       // h state (f16)
    + (size_t)BM * HID * 2           // y_prev (f16)
    + (size_t)BM * NGATE * 4         // z gate buffer (f32)
    + (size_t)6 * NGATE * 4          // biases
    + (size_t)2 * NGATE * 4          // wxE, wxD
    + (size_t)HID * 4                // fcW
    + (size_t)BM * SEQ_T * 4;        // x slice (32 KB)

  dim3 grid(BATCH / BM);             // 32 persistent workgroups
  dim3 block(NTHREADS);              // 8 wave32 waves
  seq2seq_lstm_kernel<<<grid, block, SMEM_BYTES, stream>>>(
      enc_x, dec_x,
      eWih0, eWhh0, eB0, eWih1, eWhh1, eB1, eWih2, eWhh2, eB2,
      dWih0, dWhh0, dB0, dWih1, dWhh1, dB1, dWih2, dWhh2, dB2,
      fcW, fcB, out);
}
